// STARLINE_38895223832629
// MI455X (gfx1250) — compile-verified, matrix operand order checked by reading, not run on hardware
//
#include <hip/hip_runtime.h>
#include <hip/hip_bf16.h>

#define N_USERS 8192
#define N_ITEMS 8192
#define DIM     64
#define E_EDGE  524288
#define K_SEL   262144
#define NBINS   4096
#define TAU_INV 10.0f
#define EPSG    1e-10f

typedef _Float16 v16h __attribute__((ext_vector_type(16)));
typedef _Float16 v8h  __attribute__((ext_vector_type(8)));
typedef float    v8f  __attribute__((ext_vector_type(8)));

// ---------------------------------------------------------------------------
// Row-normalize a (8192 x 64) f32 matrix into f16.  One wave per row.
// ---------------------------------------------------------------------------
__global__ void k_normalize(const float* __restrict__ src, _Float16* __restrict__ dst) {
    const int row = blockIdx.x;
    const int t   = threadIdx.x;              // 0..31
    const float* r = src + (size_t)row * DIM;
    float a0 = r[t];
    float a1 = r[t + 32];
    float s  = a0 * a0 + a1 * a1;
    #pragma unroll
    for (int m = 16; m > 0; m >>= 1) s += __shfl_xor(s, m, 32);
    const float rn = rsqrtf(s);
    dst[(size_t)row * DIM + t]      = (_Float16)(a0 * rn);
    dst[(size_t)row * DIM + t + 32] = (_Float16)(a1 * rn);
}

// ---------------------------------------------------------------------------
// Scatter existing edges into an 8192x8192 bitmask (8 MB).
// ---------------------------------------------------------------------------
__global__ void k_mask(const int* __restrict__ nz_u, const int* __restrict__ nz_i,
                       unsigned* __restrict__ mask) {
    const int e = blockIdx.x * blockDim.x + threadIdx.x;
    if (e >= E_EDGE) return;
    const unsigned idx = (unsigned)nz_u[e] * (unsigned)N_ITEMS + (unsigned)nz_i[e];
    atomicOr(&mask[idx >> 5], 1u << (idx & 31));
}

// ---------------------------------------------------------------------------
// One wave computes a 16x16 cos tile: C = A(16x64) * B(64x16) via two
// v_wmma_f32_16x16x32_f16.  A operand per documented 16-bit A layout:
// lane m = lane&15, K chunks {k0..k0+7, k0+16..k0+23}, k0 = lane<16 ? 0 : 8.
// B symmetric (n = lane&15) read from an LDS-staged en tile.
// ---------------------------------------------------------------------------
__device__ __forceinline__ v8f wmma_tile(const _Float16* __restrict__ pn,
                                         const _Float16* __restrict__ bt,
                                         int row0, int lane) {
    const int m  = lane & 15;
    const int k0 = (lane < 16) ? 0 : 8;
    const _Float16* arow = pn + (size_t)(row0 + m) * DIM;
    const _Float16* brow = bt + m * DIM;
    v8f c = {0.f, 0.f, 0.f, 0.f, 0.f, 0.f, 0.f, 0.f};
    #pragma unroll
    for (int kk = 0; kk < DIM; kk += 32) {
        v8h alo = *(const v8h*)(arow + kk + k0);
        v8h ahi = *(const v8h*)(arow + kk + k0 + 16);
        v8h blo = *(const v8h*)(brow + kk + k0);
        v8h bhi = *(const v8h*)(brow + kk + k0 + 16);
        v16h a = __builtin_shufflevector(alo, ahi, 0,1,2,3,4,5,6,7,8,9,10,11,12,13,14,15);
        v16h b = __builtin_shufflevector(blo, bhi, 0,1,2,3,4,5,6,7,8,9,10,11,12,13,14,15);
        c = __builtin_amdgcn_wmma_f32_16x16x32_f16(false, a, false, b, (short)0, c,
                                                   false, false);
    }
    return c;
}

// ---------------------------------------------------------------------------
// Pass 1: WMMA tiles -> LDS histogram of masked, positive cos values.
// Block = 256 threads = 8 waves; blockIdx.x covers 8 tile-rows, blockIdx.y one
// tile-col whose B tile is staged once in LDS.
// ---------------------------------------------------------------------------
__global__ void k_hist(const _Float16* __restrict__ pn, const _Float16* __restrict__ en,
                       const unsigned* __restrict__ mask, unsigned* __restrict__ hist) {
    __shared__ __align__(16) _Float16 bt[16 * DIM];
    __shared__ unsigned lh[NBINS];
    for (int i = threadIdx.x; i < NBINS; i += blockDim.x) lh[i] = 0u;
    const int tn0 = blockIdx.y * 16;
    {   // 16 consecutive rows of en are contiguous: 2048 B cooperative copy
        const unsigned long long* s = (const unsigned long long*)(en + (size_t)tn0 * DIM);
        unsigned long long* d = (unsigned long long*)bt;
        for (int i = threadIdx.x; i < (16 * DIM) / 4; i += blockDim.x) d[i] = s[i];
    }
    __syncthreads();

    const int lane = threadIdx.x & 31;
    const int wave = threadIdx.x >> 5;
    const int tm   = blockIdx.x * 8 + wave;
    v8f c = wmma_tile(pn, bt, tm * 16, lane);

    const int n     = lane & 15;
    const int mbase = (lane < 16) ? 0 : 8;
    const unsigned col = (unsigned)(tn0 + n);
    #pragma unroll
    for (int v = 0; v < 8; ++v) {
        const float val = c[v];
        const unsigned row  = (unsigned)(tm * 16 + mbase + v);
        const unsigned fidx = row * (unsigned)N_ITEMS + col;
        const bool masked = (mask[fidx >> 5] >> (fidx & 31)) & 1u;
        if (!masked && val > 0.0f) {
            int b = (int)(val * (float)NBINS);
            if (b > NBINS - 1) b = NBINS - 1;
            atomicAdd(&lh[b], 1u);
        }
    }
    __syncthreads();
    for (int i = threadIdx.x; i < NBINS; i += blockDim.x)
        if (lh[i]) atomicAdd(&hist[i], lh[i]);
}

// ---------------------------------------------------------------------------
// Suffix-scan the histogram (top bin down): base[b] = #values in bins > b,
// threshold bin T = largest b with suffix count >= K.
// ---------------------------------------------------------------------------
__global__ void k_scan(const unsigned* __restrict__ hist, unsigned* __restrict__ base,
                       int* __restrict__ thr) {
    if (threadIdx.x != 0 || blockIdx.x != 0) return;
    unsigned cum = 0u;
    int T = 0;
    bool found = false;
    for (int b = NBINS - 1; b >= 0; --b) {
        base[b] = cum;
        cum += hist[b];
        if (!found && cum >= (unsigned)K_SEL) { T = b; found = true; }
    }
    *thr = T;
}

// ---------------------------------------------------------------------------
// Pass 2: recompute WMMA tiles, append values in bins >= T at bin-sorted
// positions (base[b] + per-bin counter), capped at K.
// ---------------------------------------------------------------------------
__global__ void k_select(const _Float16* __restrict__ pn, const _Float16* __restrict__ en,
                         const unsigned* __restrict__ mask,
                         const unsigned* __restrict__ base, const int* __restrict__ thr,
                         unsigned* __restrict__ binctr,
                         float* __restrict__ cand_val, int* __restrict__ cand_idx) {
    __shared__ __align__(16) _Float16 bt[16 * DIM];
    const int tn0 = blockIdx.y * 16;
    {
        const unsigned long long* s = (const unsigned long long*)(en + (size_t)tn0 * DIM);
        unsigned long long* d = (unsigned long long*)bt;
        for (int i = threadIdx.x; i < (16 * DIM) / 4; i += blockDim.x) d[i] = s[i];
    }
    __syncthreads();

    const int T    = *thr;
    const int lane = threadIdx.x & 31;
    const int wave = threadIdx.x >> 5;
    const int tm   = blockIdx.x * 8 + wave;
    v8f c = wmma_tile(pn, bt, tm * 16, lane);

    const int n     = lane & 15;
    const int mbase = (lane < 16) ? 0 : 8;
    const unsigned col = (unsigned)(tn0 + n);
    #pragma unroll
    for (int v = 0; v < 8; ++v) {
        const float val = c[v];
        const unsigned row  = (unsigned)(tm * 16 + mbase + v);
        const unsigned fidx = row * (unsigned)N_ITEMS + col;
        const bool masked = (mask[fidx >> 5] >> (fidx & 31)) & 1u;
        if (!masked && val > 0.0f) {
            int b = (int)(val * (float)NBINS);
            if (b > NBINS - 1) b = NBINS - 1;
            if (b >= T) {
                const unsigned pos = base[b] + atomicAdd(&binctr[b], 1u);
                if (pos < (unsigned)K_SEL) {
                    cand_val[pos] = val;
                    cand_idx[pos] = (int)fidx;
                }
            }
        }
    }
}

// ---------------------------------------------------------------------------
// Gather x=[pref[u],emb[i]], 128x2 MLP, Gumbel-hard argmax, write outputs.
// One thread per row j in [0, E+K).
// ---------------------------------------------------------------------------
__global__ void k_mlp(const float* __restrict__ pref, const float* __restrict__ emb,
                      const float* __restrict__ W, const float* __restrict__ bvec,
                      const float* __restrict__ uni,
                      const int* __restrict__ nz_u, const int* __restrict__ nz_i,
                      const float* __restrict__ cand_val, const int* __restrict__ cand_idx,
                      float* __restrict__ out_w, float* __restrict__ out_wo) {
    const int j = blockIdx.x * blockDim.x + threadIdx.x;
    if (j >= E_EDGE + K_SEL) return;

    int u, it;
    float value;
    bool evalid;
    if (j < E_EDGE) {
        u = nz_u[j]; it = nz_i[j];
        value = 1.0f; evalid = true;
    } else {
        const int k = j - E_EDGE;
        const float v = cand_val[k];
        const int f = cand_idx[k];
        u  = f / N_ITEMS;
        it = f % N_ITEMS;
        evalid = (v > 0.0f);
        value  = evalid ? fmaxf(v, 0.0f) : 0.0f;
    }

    const float* xp = pref + (size_t)u * DIM;
    const float* xe = emb + (size_t)it * DIM;
    float l0 = bvec[0], l1 = bvec[1];
    #pragma unroll 8
    for (int d = 0; d < DIM; ++d) {
        const float pv = xp[d], ev = xe[d];
        l0 += pv * W[d * 2 + 0] + ev * W[(DIM + d) * 2 + 0];
        l1 += pv * W[d * 2 + 1] + ev * W[(DIM + d) * 2 + 1];
    }
    const float u0 = uni[2 * (size_t)j + 0];
    const float u1 = uni[2 * (size_t)j + 1];
    const float g0 = -logf(-logf(u0 + EPSG) + EPSG);
    const float g1 = -logf(-logf(u1 + EPSG) + EPSG);
    const float z0 = (l0 + g0) * TAU_INV;
    const float z1 = (l1 + g1) * TAU_INV;
    float retain = (z0 >= z1) ? 1.0f : 0.0f;   // argmax==0 (first-max tie break)
    if (!evalid) retain = 0.0f;
    const float rw = retain * value;

    out_w[j]                    = rw;
    out_w[j + E_EDGE + K_SEL]   = rw;
    out_wo[j]                   = retain;
    out_wo[j + E_EDGE + K_SEL]  = retain;
}

// ---------------------------------------------------------------------------
extern "C" void kernel_launch(void* const* d_in, const int* in_sizes, int n_in,
                              void* d_out, int out_size, void* d_ws, size_t ws_size,
                              hipStream_t stream) {
    const float* img_pref = (const float*)d_in[0];
    const float* txt_pref = (const float*)d_in[1];
    const float* img_emb  = (const float*)d_in[2];
    const float* txt_emb  = (const float*)d_in[3];
    const float* W_v      = (const float*)d_in[4];
    const float* b_v      = (const float*)d_in[5];
    const float* W_t      = (const float*)d_in[6];
    const float* b_t      = (const float*)d_in[7];
    const float* uni_v    = (const float*)d_in[8];
    const float* uni_t    = (const float*)d_in[9];
    const int*   nz_u     = (const int*)d_in[10];
    const int*   nz_i     = (const int*)d_in[11];
    float* out = (float*)d_out;

    // ---- workspace carve-up (256B aligned) ----
    char* ws = (char*)d_ws;
    size_t o = 0;
    auto take = [&](size_t n) -> char* {
        char* p = ws + o;
        o += (n + 255) & ~(size_t)255;
        return p;
    };
    _Float16* pn_v = (_Float16*)take((size_t)N_USERS * DIM * 2);
    _Float16* en_v = (_Float16*)take((size_t)N_ITEMS * DIM * 2);
    _Float16* pn_t = (_Float16*)take((size_t)N_USERS * DIM * 2);
    _Float16* en_t = (_Float16*)take((size_t)N_ITEMS * DIM * 2);
    unsigned* mask = (unsigned*)take((size_t)N_USERS * N_ITEMS / 8);      // 8 MB
    unsigned* hist_v   = (unsigned*)take(NBINS * 4);
    unsigned* hist_t   = (unsigned*)take(NBINS * 4);
    unsigned* binctr_v = (unsigned*)take(NBINS * 4);
    unsigned* binctr_t = (unsigned*)take(NBINS * 4);
    unsigned* base_v   = (unsigned*)take(NBINS * 4);
    unsigned* base_t   = (unsigned*)take(NBINS * 4);
    int*      thr_v    = (int*)take(256);
    int*      thr_t    = (int*)take(256);
    float* cand_val_v  = (float*)take((size_t)K_SEL * 4);
    int*   cand_idx_v  = (int*)take((size_t)K_SEL * 4);
    float* cand_val_t  = (float*)take((size_t)K_SEL * 4);
    int*   cand_idx_t  = (int*)take((size_t)K_SEL * 4);

    // ---- re-init scratch every call (graph-capturable memsets) ----
    hipMemsetAsync(mask, 0, (size_t)N_USERS * N_ITEMS / 8, stream);
    hipMemsetAsync(hist_v, 0, NBINS * 4, stream);
    hipMemsetAsync(hist_t, 0, NBINS * 4, stream);
    hipMemsetAsync(binctr_v, 0, NBINS * 4, stream);
    hipMemsetAsync(binctr_t, 0, NBINS * 4, stream);
    hipMemsetAsync(cand_val_v, 0, (size_t)K_SEL * 4, stream);
    hipMemsetAsync(cand_idx_v, 0, (size_t)K_SEL * 4, stream);
    hipMemsetAsync(cand_val_t, 0, (size_t)K_SEL * 4, stream);
    hipMemsetAsync(cand_idx_t, 0, (size_t)K_SEL * 4, stream);

    // ---- normalize + f16 convert ----
    k_normalize<<<N_USERS, 32, 0, stream>>>(img_pref, pn_v);
    k_normalize<<<N_ITEMS, 32, 0, stream>>>(img_emb, en_v);
    k_normalize<<<N_USERS, 32, 0, stream>>>(txt_pref, pn_t);
    k_normalize<<<N_ITEMS, 32, 0, stream>>>(txt_emb, en_t);

    // ---- edge bitmask ----
    k_mask<<<E_EDGE / 256, 256, 0, stream>>>(nz_u, nz_i, mask);

    const dim3 tgrid(N_USERS / (16 * 8), N_ITEMS / 16);   // (64, 512), 8 waves/block
    const int  mlp_blocks = (E_EDGE + K_SEL + 255) / 256;
    const size_t L = 2 * (size_t)(E_EDGE + K_SEL);

    // ---- image modality ----
    k_hist<<<tgrid, 256, 0, stream>>>(pn_v, en_v, mask, hist_v);
    k_scan<<<1, 1, 0, stream>>>(hist_v, base_v, thr_v);
    k_select<<<tgrid, 256, 0, stream>>>(pn_v, en_v, mask, base_v, thr_v, binctr_v,
                                        cand_val_v, cand_idx_v);
    k_mlp<<<mlp_blocks, 256, 0, stream>>>(img_pref, img_emb, W_v, b_v, uni_v,
                                          nz_u, nz_i, cand_val_v, cand_idx_v,
                                          out + 0 * L, out + 2 * L);

    // ---- text modality ----
    k_hist<<<tgrid, 256, 0, stream>>>(pn_t, en_t, mask, hist_t);
    k_scan<<<1, 1, 0, stream>>>(hist_t, base_t, thr_t);
    k_select<<<tgrid, 256, 0, stream>>>(pn_t, en_t, mask, base_t, thr_t, binctr_t,
                                        cand_val_t, cand_idx_t);
    k_mlp<<<mlp_blocks, 256, 0, stream>>>(txt_pref, txt_emb, W_t, b_t, uni_t,
                                          nz_u, nz_i, cand_val_t, cand_idx_t,
                                          out + 1 * L, out + 3 * L);
}
